// MILBase_48249662603644
// MI455X (gfx1250) — matrix operand order repaired and checked by reading, not compiled, not run on hardware
//
#include <hip/hip_runtime.h>
#include <hip/hip_bf16.h>
#include <float.h>

// ---------------------------------------------------------------------------
// Gated-attention MIL pooling for MI455X (gfx1250, wave32, WMMA bf16 path).
//   a = tanh(x@Wa), g = sigmoid(x@Wb), A = (a*g)@Wc  -> masked softmax over N
//   out[b,:] = sum_n softmax(A)[b,n] * x[b,n,:]
// B=8, N=8192, L=1024, D=128. Memory-bound (~536MB of x traffic @ 23.3TB/s);
// bf16 WMMA keeps matrix math far above the HBM roofline.
// ---------------------------------------------------------------------------

typedef __attribute__((ext_vector_type(16))) __bf16 v16bf;
typedef __attribute__((ext_vector_type(8)))  __bf16 v8bf;
typedef __attribute__((ext_vector_type(8)))  float  v8f;
typedef __attribute__((ext_vector_type(4)))  float  v4f;

constexpr int B_ = 8;
constexpr int N_ = 8192;
constexpr int L_ = 1024;   // K dimension of the GEMMs
constexpr int D_ = 128;
constexpr int COLS = 2 * D_;        // Wa|Wb fused -> 256 output columns
constexpr int MT   = 64;            // rows of x per workgroup (4 row-tiles of 16)
constexpr int KT   = 32;            // K per WMMA step (bf16 16x16x32)
constexpr int XS_STRIDE = KT + 8;   // padded LDS row stride (80B, 16B-aligned)

constexpr int POOL_NCHUNK = 512;
constexpr int POOL_CL = L_ / 256;          // 4
constexpr int POOL_CN = N_ / POOL_NCHUNK;  // 16

// Workspace layout (d_ws): [Wt bf16 256x1024 | logits f32 B*N | weights f32 B*N]
constexpr size_t WS_WT_BYTES     = (size_t)COLS * L_ * sizeof(__bf16); // 512 KB
constexpr size_t WS_LOGIT_BYTES  = (size_t)B_ * N_ * sizeof(float);    // 256 KB
constexpr size_t WS_WT_OFF       = 0;
constexpr size_t WS_LOGIT_OFF    = WS_WT_OFF + WS_WT_BYTES;
constexpr size_t WS_WGT_OFF      = WS_LOGIT_OFF + WS_LOGIT_BYTES;

// ---------------------------------------------------------------------------
__global__ void zero_k(float* p, int n) {
    int i = blockIdx.x * 256 + threadIdx.x;
    if (i < n) p[i] = 0.0f;
}

// Pack Wa||Wb into bf16, transposed: Wt[c][k], c in [0,256).  Column-major per
// output column so a B-fragment (16 contiguous K values per lane) is one
// contiguous 32B load -> k = 16*(lane/16) + s, the same order the A fragments use.
__global__ void prep_wt(const float* __restrict__ Wa, const float* __restrict__ Wb,
                        __bf16* __restrict__ Wt) {
    int idx = blockIdx.x * 256 + threadIdx.x;        // COLS*L_ total
    int c = idx / L_;
    int k = idx - c * L_;
    const float* W = (c < D_) ? Wa : Wb;
    int d = c & (D_ - 1);
    Wt[(size_t)c * L_ + k] = (__bf16)W[(size_t)k * D_ + d];
}

// ---------------------------------------------------------------------------
// Pass 1: logits[b,n] = (tanh(xWa) * sigmoid(xWb)) @ Wc, masked.
// 8 waves/WG. WG owns 64 rows x 256 cols. Wave w owns col-tiles {w, w+8}
// (same d for the tanh half and the sigmoid half -> local gating epilogue).
__global__ __launch_bounds__(256) void gate_logits(
        const float* __restrict__ x, const unsigned char* __restrict__ mask,
        const __bf16* __restrict__ Wt, const float* __restrict__ Wc,
        float* __restrict__ logits) {
    __shared__ __align__(16) __bf16 xs[MT * XS_STRIDE];   // 5 KB bf16 x tile
    __shared__ float logit_s[MT];

    const int tid  = threadIdx.x;
    const int lane = tid & 31;
    const int wave = tid >> 5;

    const size_t R0   = (size_t)blockIdx.x * MT;   // global row base
    const int    b    = (int)(R0 / N_);
    const int    n0   = (int)(R0 % N_);            // MT | N_, so rows stay in one bag
    const size_t xoff = R0 * L_;

    // Accumulators: acc[mt][0]=tanh-half (col-tile wave), acc[mt][1]=sig-half.
    v8f acc[4][2];
#pragma unroll
    for (int mt = 0; mt < 4; ++mt)
#pragma unroll
        for (int p = 0; p < 2; ++p) acc[mt][p] = (v8f)0.0f;

    // A-fragment addressing: lanes 0-15 -> K {0..15},
    // lanes 16-31 -> K {16..31}; row = lane%16 within the 16-row tile.
    const int s_a   = (lane < 16) ? 0 : 16;
    const int row_a = lane & 15;
    // B-fragment addressing: lane%16 = output column, lanes>=16 take K+16.
    const int colA = wave * 16 + (lane & 15);        // d in [0,128)
    const int s_b  = (lane < 16) ? 0 : 16;
    const __bf16* wAp = Wt + (size_t)colA * L_ + s_b;          // tanh half
    const __bf16* wGp = Wt + (size_t)(colA + D_) * L_ + s_b;   // sigmoid half

    // Cooperative staging map: 256 threads x 8 floats = 64 rows x 32 K.
    const int st_row = tid >> 2;
    const int st_off = (tid & 3) * 8;
    const float* st_ptr = x + xoff + (size_t)st_row * L_ + st_off;

    for (int k0 = 0; k0 < L_; k0 += KT) {
        __syncthreads();   // protect xs from previous iteration's readers
        {
            const float* xp = st_ptr + k0;
            if (k0 + KT < L_) __builtin_prefetch(xp + KT, 0, 3);
            v4f f0 = *(const v4f*)xp;
            v4f f1 = *(const v4f*)(xp + 4);
            v8bf h;
#pragma unroll
            for (int i = 0; i < 4; ++i) {
                h[i]     = (__bf16)f0[i];
                h[4 + i] = (__bf16)f1[i];
            }
            *(v8bf*)&xs[st_row * XS_STRIDE + st_off] = h;
        }
        __syncthreads();

        // Load 4 A fragments (one per 16-row tile) from LDS.
        v16bf afr[4];
#pragma unroll
        for (int mt = 0; mt < 4; ++mt) {
            const __bf16* ap = &xs[(mt * 16 + row_a) * XS_STRIDE + s_a];
            v8bf lo = *(const v8bf*)ap;          // K s_a .. s_a+7
            v8bf hi = *(const v8bf*)(ap + 8);    // K s_a+8 .. s_a+15
#pragma unroll
            for (int i = 0; i < 8; ++i) { afr[mt][i] = lo[i]; afr[mt][8 + i] = hi[i]; }
        }

        // tanh-half B fragment, 4 WMMAs (reused across the 4 row tiles).
        {
            v8bf b0 = *(const v8bf*)(wAp + k0);
            v8bf b1 = *(const v8bf*)(wAp + k0 + 8);
            v16bf bf;
#pragma unroll
            for (int i = 0; i < 8; ++i) { bf[i] = b0[i]; bf[8 + i] = b1[i]; }
#pragma unroll
            for (int mt = 0; mt < 4; ++mt)
                acc[mt][0] = __builtin_amdgcn_wmma_f32_16x16x32_bf16(
                    false, afr[mt], false, bf, (short)0, acc[mt][0], false, false);
        }
        // sigmoid-half B fragment, 4 WMMAs.
        {
            v8bf b0 = *(const v8bf*)(wGp + k0);
            v8bf b1 = *(const v8bf*)(wGp + k0 + 8);
            v16bf bf;
#pragma unroll
            for (int i = 0; i < 8; ++i) { bf[i] = b0[i]; bf[8 + i] = b1[i]; }
#pragma unroll
            for (int mt = 0; mt < 4; ++mt)
                acc[mt][1] = __builtin_amdgcn_wmma_f32_16x16x32_bf16(
                    false, afr[mt], false, bf, (short)0, acc[mt][1], false, false);
        }
    }

    // Epilogue: h = tanh(a)*sigmoid(g)*Wc[d]; reduce over d (16 lanes in-wave,
    // 8 waves via LDS float atomics), then mask + store the logit.
    if (tid < MT) logit_s[tid] = 0.0f;
    __syncthreads();

    const float wc = Wc[colA];
#pragma unroll
    for (int mt = 0; mt < 4; ++mt) {
#pragma unroll
        for (int r = 0; r < 8; ++r) {
            // C/D layout: lane<16 -> row r, lane>=16 -> row r+8, col = lane%16.
            float a = tanhf(acc[mt][0][r]);
            float g = 1.0f / (1.0f + __expf(-acc[mt][1][r]));
            float h = a * g * wc;
            h += __shfl_xor(h, 1, 32);
            h += __shfl_xor(h, 2, 32);
            h += __shfl_xor(h, 4, 32);
            h += __shfl_xor(h, 8, 32);   // masks <16 keep the two halves separate
            if ((lane & 15) == 0) {
                int row = mt * 16 + r + ((lane < 16) ? 0 : 8);
                atomicAdd(&logit_s[row], h);   // ds_add_f32
            }
        }
    }
    __syncthreads();

    if (tid < MT) {
        size_t gi = (size_t)b * N_ + (n0 + tid);
        logits[gi] = mask[gi] ? -FLT_MAX : logit_s[tid];
    }
}

// ---------------------------------------------------------------------------
// Pass 2: per-bag softmax over N. One 1024-thread (32-wave) WG per bag.
__global__ __launch_bounds__(1024) void softmax_k(const float* __restrict__ logits,
                                                  float* __restrict__ wgt) {
    const int b = blockIdx.x;
    const int tid = threadIdx.x, lane = tid & 31, wid = tid >> 5;
    __shared__ float red[32];
    __shared__ float bmax, bsum;
    const float* lp = logits + (size_t)b * N_;

    float m = -FLT_MAX;
    for (int n = tid; n < N_; n += 1024) m = fmaxf(m, lp[n]);
#pragma unroll
    for (int o = 16; o >= 1; o >>= 1) m = fmaxf(m, __shfl_xor(m, o, 32));
    if (lane == 0) red[wid] = m;
    __syncthreads();
    if (wid == 0) {
        float v = red[lane];
#pragma unroll
        for (int o = 16; o >= 1; o >>= 1) v = fmaxf(v, __shfl_xor(v, o, 32));
        if (lane == 0) bmax = v;
    }
    __syncthreads();
    m = bmax;

    float s = 0.0f;
    for (int n = tid; n < N_; n += 1024) s += __expf(lp[n] - m);
#pragma unroll
    for (int o = 16; o >= 1; o >>= 1) s += __shfl_xor(s, o, 32);
    __syncthreads();
    if (lane == 0) red[wid] = s;
    __syncthreads();
    if (wid == 0) {
        float v = red[lane];
#pragma unroll
        for (int o = 16; o >= 1; o >>= 1) v += __shfl_xor(v, o, 32);
        if (lane == 0) bsum = v;
    }
    __syncthreads();

    const float inv = 1.0f / bsum;
    float* wp = wgt + (size_t)b * N_;
    for (int n = tid; n < N_; n += 1024) wp[n] = __expf(lp[n] - m) * inv;
}

// ---------------------------------------------------------------------------
// Pass 3: out[b,l] = sum_n w[b,n] * x[b,n,l].  N split for parallelism,
// weights staged in LDS, fp32 global atomics for the partial sums.
__global__ __launch_bounds__(256) void pool_k(const float* __restrict__ x,
                                              const float* __restrict__ wgt,
                                              float* __restrict__ out) {
    const int bid = blockIdx.x;
    const int b   = bid / (POOL_CL * POOL_CN);
    const int rem = bid % (POOL_CL * POOL_CN);
    const int lc  = rem / POOL_CN;
    const int nc  = rem % POOL_CN;
    const int l   = lc * 256 + threadIdx.x;
    const int nb  = nc * POOL_NCHUNK;

    __shared__ float wsh[POOL_NCHUNK];
    for (int i = threadIdx.x; i < POOL_NCHUNK; i += 256)
        wsh[i] = wgt[(size_t)b * N_ + nb + i];
    __syncthreads();

    const float* xp = x + ((size_t)b * N_ + nb) * L_ + l;
    float acc = 0.0f;
#pragma unroll 8
    for (int i = 0; i < POOL_NCHUNK; ++i) acc += wsh[i] * xp[(size_t)i * L_];
    atomicAdd(&out[(size_t)b * L_ + l], acc);
}

// ---------------------------------------------------------------------------
extern "C" void kernel_launch(void* const* d_in, const int* in_sizes, int n_in,
                              void* d_out, int out_size, void* d_ws, size_t ws_size,
                              hipStream_t stream) {
    (void)in_sizes; (void)n_in; (void)out_size; (void)ws_size;
    const float*         x    = (const float*)d_in[0];
    const unsigned char* mask = (const unsigned char*)d_in[1];  // jnp bool_ = 1 byte
    const float*         Wa   = (const float*)d_in[2];
    const float*         Wb   = (const float*)d_in[3];
    const float*         Wc   = (const float*)d_in[4];
    float*               out  = (float*)d_out;

    char*    ws     = (char*)d_ws;
    __bf16*  Wt     = (__bf16*)(ws + WS_WT_OFF);
    float*   logits = (float*)(ws + WS_LOGIT_OFF);
    float*   wgt    = (float*)(ws + WS_WGT_OFF);

    zero_k<<<(B_ * L_ + 255) / 256, 256, 0, stream>>>(out, B_ * L_);
    prep_wt<<<(COLS * L_) / 256, 256, 0, stream>>>(Wa, Wb, Wt);
    gate_logits<<<(B_ * N_) / MT, 256, 0, stream>>>(x, mask, Wt, Wc, logits);
    softmax_k<<<B_, 1024, 0, stream>>>(logits, wgt);
    pool_k<<<B_ * POOL_CL * POOL_CN, 256, 0, stream>>>(x, wgt, out);
}
